// NEGAN_6451040878845
// MI455X (gfx1250) — matrix operand order repaired and checked by reading, not compiled, not run on hardware
//
#include <hip/hip_runtime.h>
#include <hip/hip_bf16.h>
#include <math.h>

typedef __bf16 bf16_t;
typedef __attribute__((ext_vector_type(16))) __bf16 v16bf;
typedef __attribute__((ext_vector_type(8)))  float  v8f;

__device__ __forceinline__ v8f zero8() {
    v8f v;
#pragma unroll
    for (int i = 0; i < 8; ++i) v[i] = 0.f;
    return v;
}

// ---------------------------------------------------------------------------
// Generic batched GEMM via V_WMMA_F32_16X16X32_BF16 (wave32).
// C[b] = act(alpha * A[b] (MxK) * B[b] (KxN) + bias)
// Block = 128 threads (4 waves), 64x64 C tile; wave w owns a 16x64 strip
// (4 accumulators reuse one A fragment). B tile (32x64) is staged per K-chunk
// into LDS bf16 (row padded to 40 elems -> conflict-free 16B-aligned b128
// ds traffic) and shared by all 4 waves. All 4 B fragments are preloaded
// before the WMMA chain so ds waits pipeline instead of fully serializing.
// AK1/BK1: compile-time "k-stride == 1" specialization (contiguous runs merge
// into global_load_b128; all offsets fit int32 -> no 64-bit multiplies).
// Edge tiles: clamped indices + multiply-by-{0,1} masks, with loads hoisted
// into a separate loop so they clause up under a single wait.
// ACT: 0 = none, 1 = relu, 2 = leaky_relu(0.2)
// ---------------------------------------------------------------------------
template <int ACT, int AK1, int BK1>
__global__ __launch_bounds__(128) void wmma_gemm(
    const float* __restrict__ Ag, const float* __restrict__ Bg,
    float* __restrict__ Cg, const float* __restrict__ bias, float alpha,
    int M, int N, int K,
    int sAm, int sAk_, int sBk_, int sBn, int sCm,
    int bA, int bB, int bC)
{
    __shared__ __bf16 Bs[64][40];              // [n][k], padded row

    const int sAk = AK1 ? 1 : sAk_;
    const int sBk = BK1 ? 1 : sBk_;

    const int tid  = threadIdx.x;
    const int lane = tid & 31;
    const int wave = tid >> 5;
    const int m0   = blockIdx.x * 64 + wave * 16;
    const int n0   = blockIdx.y * 64;
    const int z    = blockIdx.z;

    const float* A = Ag + (long)z * bA;
    const float* B = Bg + (long)z * bB;
    float*       C = Cg + (long)z * bC;

    const int r = lane & 15;                   // row/col within 16x16 tile
    const int h = lane >> 4;                   // lane half

    v8f acc[4];
#pragma unroll
    for (int u = 0; u < 4; ++u) acc[u] = zero8();

    const int  mRow  = m0 + r;
    const bool mOk   = (mRow < M);
    const bool fullM = (m0 + 16 <= M);         // wave-uniform
    const int  aRow  = (mOk ? mRow : 0) * sAm;
    const float mMask = mOk ? 1.f : 0.f;

    // B staging map: 128 threads cover 64 tile columns x two 16-long k runs
    const int  bn    = tid >> 1;               // 0..63  (tile column index)
    const int  bk0   = (tid & 1) * 16;         // 0 or 16 (k start within chunk)
    const int  gn    = n0 + bn;
    const bool nOk   = (gn < N);
    const bool fullN = (n0 + 64 <= N);         // block-uniform
    const int  bCol  = (nOk ? gn : 0) * sBn;
    const float nMask = nOk ? 1.f : 0.f;

    for (int kk = 0; kk < K; kk += 32) {
        const bool fullK = (kk + 32 <= K);     // uniform

        // ---- cooperative stage of B tile into LDS -------------------------
        if (fullN && fullK) {
            if (BK1) {
                const float* src = B + bCol + kk + bk0;     // contiguous run
#pragma unroll
                for (int j = 0; j < 16; ++j)
                    Bs[bn][bk0 + j] = (bf16_t)src[j];
            } else {
#pragma unroll
                for (int j = 0; j < 16; ++j)
                    Bs[bn][bk0 + j] = (bf16_t)B[(kk + bk0 + j) * sBk + bCol];
            }
        } else {
            float vb[16];
#pragma unroll
            for (int j = 0; j < 16; ++j) {      // pure loads: clause + 1 wait
                const int k = kk + bk0 + j;
                vb[j] = B[(k < K ? k : 0) * sBk + bCol];
            }
#pragma unroll
            for (int j = 0; j < 16; ++j) {      // mask + convert + store
                const int   k = kk + bk0 + j;
                const float f = (k < K) ? nMask : 0.f;
                Bs[bn][bk0 + j] = (bf16_t)(vb[j] * f);
            }
        }
        __syncthreads();

        if (kk + 32 < K)                       // speculative prefetch of next A chunk
            __builtin_prefetch(A + aRow + (kk + 32) * sAk, 0, 1);

        // ---- A fragment: 16x32 bf16, direct from global -------------------
        v16bf a;
        if (fullM && fullK) {                  // unconditional, vectorizable
            if (AK1) {
                const float* pa = A + aRow + kk;
#pragma unroll
                for (int j = 0; j < 8; ++j) {
                    a[j]     = (bf16_t)pa[h * 8 + j];
                    a[j + 8] = (bf16_t)pa[16 + h * 8 + j];
                }
            } else {
#pragma unroll
                for (int j = 0; j < 8; ++j) {
                    a[j]     = (bf16_t)A[aRow + (kk + h * 8 + j) * sAk];
                    a[j + 8] = (bf16_t)A[aRow + (kk + 16 + h * 8 + j) * sAk];
                }
            }
        } else {                               // clamped; loads first, then masks
            float v1[8], v2[8];
#pragma unroll
            for (int j = 0; j < 8; ++j) {
                const int k1 = kk + h * 8 + j;
                const int k2 = k1 + 16;
                v1[j] = A[aRow + (k1 < K ? k1 : 0) * sAk];
                v2[j] = A[aRow + (k2 < K ? k2 : 0) * sAk];
            }
#pragma unroll
            for (int j = 0; j < 8; ++j) {
                const int k1 = kk + h * 8 + j;
                const int k2 = k1 + 16;
                a[j]     = (bf16_t)(v1[j] * ((k1 < K) ? mMask : 0.f));
                a[j + 8] = (bf16_t)(v2[j] * ((k2 < K) ? mMask : 0.f));
            }
        }

        // ---- preload all 4 B fragments, then 4 WMMAs ----------------------
        v16bf bf[4];
#pragma unroll
        for (int u = 0; u < 4; ++u)
#pragma unroll
            for (int t = 0; t < 16; ++t)
                bf[u][t] = Bs[u * 16 + r][h * 8 + (t & 7) + (t >> 3) * 16];
#pragma unroll
        for (int u = 0; u < 4; ++u)
            acc[u] = __builtin_amdgcn_wmma_f32_16x16x32_bf16(
                false, a, false, bf[u], (short)0, acc[u], false, false);
        __syncthreads();
    }

#pragma unroll
    for (int u = 0; u < 4; ++u) {
        const int col = n0 + u * 16 + r;
        if (col >= N) continue;
        const float bv = bias ? bias[col] : 0.f;
#pragma unroll
        for (int i = 0; i < 8; ++i) {
            const int row = m0 + h * 8 + i;
            if (row < M) {
                float v = acc[u][i] * alpha + bv;
                if (ACT == 1) v = fmaxf(v, 0.f);
                if (ACT == 2) v = (v > 0.f) ? v : 0.2f * v;
                C[row * sCm + col] = v;
            }
        }
    }
}

// ---------------------------------------------------------------------------
__global__ void fill_zero(float* __restrict__ p, long n)
{
    long i  = (long)blockIdx.x * blockDim.x + threadIdx.x;
    long st = (long)gridDim.x * blockDim.x;
    for (; i < n; i += st) p[i] = 0.f;
}

// BatchNorm1d training stats over R rows of an (R,128) matrix; one block/col.
__global__ __launch_bounds__(256) void bn_stats(
    const float* __restrict__ H, int R,
    float* __restrict__ mean, float* __restrict__ rstd)
{
    __shared__ float s1[256], s2[256];
    const int c = blockIdx.x;
    const int t = threadIdx.x;
    float a = 0.f, b = 0.f;
    for (int r = t; r < R; r += 256) {
        const float v = H[(long)r * 128 + c];
        a += v; b += v * v;
    }
    s1[t] = a; s2[t] = b; __syncthreads();
    for (int s = 128; s > 0; s >>= 1) {
        if (t < s) { s1[t] += s1[t + s]; s2[t] += s2[t + s]; }
        __syncthreads();
    }
    if (t == 0) {
        const float m = s1[0] / (float)R;
        const float v = s2[0] / (float)R - m * m;
        mean[c] = m;
        rstd[c] = rsqrtf(v + 1e-5f);
    }
}

// Apply BN + scatter into the padded XZ concat buffer at channel colOff.
__global__ void bn_apply(const float* __restrict__ H,
                         const float* __restrict__ mean, const float* __restrict__ rstd,
                         const float* __restrict__ g, const float* __restrict__ beta,
                         float* __restrict__ XZ, int n_i, int colOff, long total)
{
    long i  = (long)blockIdx.x * blockDim.x + threadIdx.x;
    long st = (long)gridDim.x * blockDim.x;
    for (; i < total; i += st) {
        const int  c  = (int)(i & 127);
        const long r  = i >> 7;
        const int  b  = (int)(r / n_i);
        const int  rr = (int)(r - (long)b * n_i);
        const float v = g[c] * (H[i] - mean[c]) * rstd[c] + beta[c];
        XZ[((long)b * 512 + rr) * 768 + colOff + c] = v;
    }
}

// Row softmax * elementwise Aold -> Anew. One block per row.
__global__ __launch_bounds__(256) void softmax_mul(
    const float* __restrict__ SCb, const float* __restrict__ Aold,
    float* __restrict__ Anew, int n)
{
    const long row = blockIdx.x;
    const float* x  = SCb  + row * (long)n;
    const float* ao = Aold + row * (long)n;
    float*       y  = Anew + row * (long)n;
    __shared__ float red[256];
    const int t = threadIdx.x;

    float mx = -3.4e38f;
    for (int j = t; j < n; j += 256) mx = fmaxf(mx, x[j]);
    red[t] = mx; __syncthreads();
    for (int s = 128; s > 0; s >>= 1) { if (t < s) red[t] = fmaxf(red[t], red[t + s]); __syncthreads(); }
    mx = red[0]; __syncthreads();

    float sm = 0.f;
    for (int j = t; j < n; j += 256) sm += __expf(x[j] - mx);
    red[t] = sm; __syncthreads();
    for (int s = 128; s > 0; s >>= 1) { if (t < s) red[t] += red[t + s]; __syncthreads(); }
    const float inv = 1.f / red[0];

    for (int j = t; j < n; j += 256) y[j] = __expf(x[j] - mx) * inv * ao[j];
}

// Row softmax + per-row entropy. One block per row (kd <= 100).
__global__ __launch_bounds__(128) void softmax_ent(
    const float* __restrict__ S, float* __restrict__ P,
    float* __restrict__ entrow, int kd)
{
    const long row = blockIdx.x;
    const float* x = S + row * (long)kd;
    float*       p = P + row * (long)kd;
    __shared__ float red[128];
    const int t = threadIdx.x;

    float mx = -3.4e38f;
    for (int j = t; j < kd; j += 128) mx = fmaxf(mx, x[j]);
    red[t] = mx; __syncthreads();
    for (int s = 64; s > 0; s >>= 1) { if (t < s) red[t] = fmaxf(red[t], red[t + s]); __syncthreads(); }
    mx = red[0]; __syncthreads();

    float sm = 0.f;
    for (int j = t; j < kd; j += 128) sm += __expf(x[j] - mx);
    red[t] = sm; __syncthreads();
    for (int s = 64; s > 0; s >>= 1) { if (t < s) red[t] += red[t + s]; __syncthreads(); }
    const float inv = 1.f / red[0]; __syncthreads();

    float e = 0.f;
    for (int j = t; j < kd; j += 128) {
        const float pv = __expf(x[j] - mx) * inv;
        p[j] = pv;
        e -= pv * logf(pv + 1e-15f);
    }
    red[t] = e; __syncthreads();
    for (int s = 64; s > 0; s >>= 1) { if (t < s) red[t] += red[t + s]; __syncthreads(); }
    if (t == 0) entrow[row] = red[0];
}

// Fixed-order partial sum-of-squares: 256 blocks write 256 partials.
__global__ __launch_bounds__(256) void reduce_sq(
    const float* __restrict__ x, long n, float* __restrict__ part)
{
    __shared__ float red[256];
    const int t = threadIdx.x;
    float a = 0.f;
    for (long i = (long)blockIdx.x * 256 + t; i < n; i += (long)gridDim.x * 256) {
        const float v = x[i]; a += v * v;
    }
    red[t] = a; __syncthreads();
    for (int s = 128; s > 0; s >>= 1) { if (t < s) red[t] += red[t + s]; __syncthreads(); }
    if (t == 0) part[blockIdx.x] = red[0];
}

// Per-layer loss finalize (single block, fixed order => deterministic).
// link += sqrt( sum(A^2) - 2*trace(Ap) + ||s^T s||_F^2 ) / diff.size
// ent  += mean(entrow)
__global__ __launch_bounds__(256) void finalize_layer(
    const float* __restrict__ partA, const float* __restrict__ partG,
    const float* __restrict__ Ap, int bsz, int kd, long apStride,
    const float* __restrict__ entrow, long nrows,
    float sizeDiff, float* __restrict__ scal)
{
    __shared__ float red[256];
    const int t = threadIdx.x;
    float e = 0.f;
    for (long i = t; i < nrows; i += 256) e += entrow[i];
    red[t] = e; __syncthreads();
    for (int s = 128; s > 0; s >>= 1) { if (t < s) red[t] += red[t + s]; __syncthreads(); }
    if (t == 0) {
        float sA = 0.f, sG = 0.f;
        for (int i = 0; i < 256; ++i) { sA += partA[i]; sG += partG[i]; }
        float tr = 0.f;
        for (int b = 0; b < bsz; ++b)
            for (int k = 0; k < kd; ++k)
                tr += Ap[(long)b * apStride + (long)k * kd + k];
        float tot = sA - 2.f * tr + sG;
        tot = (tot > 0.f) ? tot : 0.f;
        scal[0] += sqrtf(tot) / sizeDiff;
        scal[1] += red[0] / (float)nrows;
    }
}

__global__ void write_scalars(const float* __restrict__ scal,
                              float* __restrict__ oLink, float* __restrict__ oEnt)
{
    if (threadIdx.x == 0) { oLink[0] = scal[0]; oEnt[0] = scal[1]; }
}

// ---------------------------------------------------------------------------
// Host-side helpers
// ---------------------------------------------------------------------------
static inline void gemm(hipStream_t s, int act,
                        const float* A, const float* B, float* C,
                        const float* bias, float alpha, int M, int N, int K,
                        int sAm, int sAk, int sBk, int sBn, int sCm,
                        int bA, int bB, int bC, int batch)
{
    dim3 grid((M + 63) / 64, (N + 63) / 64, batch);
    dim3 block(128);
    const int ak = (sAk == 1) ? 1 : 0;
    const int bk = (sBk == 1) ? 1 : 0;
#define GO(A_, KA_, KB_) wmma_gemm<A_, KA_, KB_><<<grid, block, 0, s>>>( \
        A, B, C, bias, alpha, M, N, K, sAm, sAk, sBk, sBn, sCm, bA, bB, bC)
    switch (act * 4 + ak * 2 + bk) {
    case 0:  GO(0, 0, 0); break;
    case 1:  GO(0, 0, 1); break;
    case 2:  GO(0, 1, 0); break;
    case 3:  GO(0, 1, 1); break;
    case 4:  GO(1, 0, 0); break;
    case 5:  GO(1, 0, 1); break;
    case 6:  GO(1, 1, 0); break;
    case 7:  GO(1, 1, 1); break;
    case 8:  GO(2, 0, 0); break;
    case 9:  GO(2, 0, 1); break;
    case 10: GO(2, 1, 0); break;
    default: GO(2, 1, 1); break;
    }
#undef GO
}

// Linear(200,128)+ReLU+BatchNorm(train) then scatter into XZ at colOff.
static inline void run_line(hipStream_t s, const float* in, int rows,
                            const float* W, const float* b,
                            const float* g, const float* beta,
                            float* HL, float* MEAN, float* RSTD,
                            float* XZ, int n_i, int colOff)
{
    gemm(s, 1, in, W, HL, b, 1.f, rows, 128, 200,
         200, 1, 1, 200, 128, 0, 0, 0, 1);
    bn_stats<<<128, 256, 0, s>>>(HL, rows, MEAN, RSTD);
    bn_apply<<<2048, 256, 0, s>>>(HL, MEAN, RSTD, g, beta, XZ, n_i, colOff,
                                  (long)rows * 128);
}

extern "C" void kernel_launch(void* const* d_in, const int* in_sizes, int n_in,
                              void* d_out, int out_size, void* d_ws, size_t ws_size,
                              hipStream_t stream)
{
    (void)n_in; (void)out_size; (void)ws_size;
    // ---- input pointer table: detect flatten scheme by first size ----------
    const bool sorted = (in_sizes[0] == 64 * 512 * 512);  // A first => sorted keys
    const float *Ain, *Xin, *Zin, *adjp;
    const float *Wq[2], *bq[2], *Wk[2], *bk[2];
    const float *wsP[2];
    const float *lnW[3], *lnb[3], *lng[3], *lnbeta[3];
    const float *leW[3], *leb[3], *leg[3], *lebeta[3];
    const float *gcnW, *gcnb, *W1, *b1, *W2, *b2;
    #define F(i) ((const float*)d_in[(i)])
    if (sorted) {
        Ain = F(0); Xin = F(1); Zin = F(2); adjp = F(3);
        for (int i = 0; i < 2; ++i) {
            Wk[i] = F(4 + 4 * i); Wq[i] = F(5 + 4 * i);
            bk[i] = F(6 + 4 * i); bq[i] = F(7 + 4 * i);
        }
        W1 = F(12); W2 = F(13); b1 = F(14); b2 = F(15);
        gcnW = F(16); gcnb = F(17);
        for (int i = 0; i < 3; ++i) {          // sorted keys: W,b,beta,g
            leW[i] = F(18 + 4 * i); leb[i] = F(19 + 4 * i);
            lebeta[i] = F(20 + 4 * i); leg[i] = F(21 + 4 * i);
            lnW[i] = F(30 + 4 * i); lnb[i] = F(31 + 4 * i);
            lnbeta[i] = F(32 + 4 * i); lng[i] = F(33 + 4 * i);
        }
        wsP[0] = F(42); wsP[1] = F(43);
    } else {                                    // insertion order
        Xin = F(0); Ain = F(1); Zin = F(2); adjp = F(3);
        for (int i = 0; i < 2; ++i) {
            Wq[i] = F(4 + 4 * i); bq[i] = F(5 + 4 * i);
            Wk[i] = F(6 + 4 * i); bk[i] = F(7 + 4 * i);
        }
        wsP[0] = F(12); wsP[1] = F(13);
        for (int i = 0; i < 3; ++i) {          // insertion: W,b,g,beta
            lnW[i] = F(14 + 4 * i); lnb[i] = F(15 + 4 * i);
            lng[i] = F(16 + 4 * i); lnbeta[i] = F(17 + 4 * i);
            leW[i] = F(26 + 4 * i); leb[i] = F(27 + 4 * i);
            leg[i] = F(28 + 4 * i); lebeta[i] = F(29 + 4 * i);
        }
        gcnW = F(38); gcnb = F(39);
        W1 = F(40); b1 = F(41); W2 = F(42); b2 = F(43);
    }
    #undef F

    // ---- output layout -----------------------------------------------------
    float* out   = (float*)d_out;
    float* oY    = out;                          // 64*512*128
    float* oLink = out + 4194304;
    float* oEnt  = out + 4194305;
    float* oFA   = out + 4194306;                // 64*50*50
    float* oS0   = out + 4194306 + 160000;       // 64*512*100
    float* oS1   = oS0 + 3276800;                // 64*100*50

    // ---- workspace carve-up (floats) --------------------------------------
    float* w = (float*)d_ws;
    float* XZ   = w; w += 64L * 512 * 768;       // concat buffer (pre-zeroed)
    float* HC   = w; w += 32768L * 1024;         // classifier hidden
    float* HL   = w; w += 32768L * 128;          // line GEMM output
    float* Qb   = w; w += 64L * 512 * 200;
    float* Kb   = w; w += 64L * 512 * 200;
    float* SC   = w; w += 64L * 512 * 512;       // raw attention scores
    float* AN   = w; w += 64L * 512 * 512;       // softmax(scores)*A
    float* X1b  = w; w += 64L * 512 * 200;
    float* SSM  = w; w += 64L * 512 * 100;       // softmax(S)
    float* XP0  = w; w += 64L * 100 * 200;
    float* ZP0  = w; w += 64L * 100 * 200;
    float* XP1  = w; w += 64L * 100 * 200;
    float* ZP1  = w; w += 64L * 100 * 200;
    float* ASb  = w; w += 64L * 512 * 100;       // A @ s
    float* APb  = w; w += 64L * 100 * 100;       // layer-0 pooled adjacency
    float* Gb   = w; w += 64L * 100 * 100;       // s^T s
    float* ZCb  = w; w += 64L * 100 * 200;       // GCN temp
    float* ENTR = w; w += 32768;
    float* PAq  = w; w += 256;
    float* PGq  = w; w += 256;
    float* MEAN = w; w += 128;
    float* RSTD = w; w += 128;
    float* SCAL = w; w += 8;

    // ---- pipeline ----------------------------------------------------------
    fill_zero<<<4096, 256, 0, stream>>>(XZ, 64L * 512 * 768);
    fill_zero<<<1, 32, 0, stream>>>(SCAL, 8);

    // layer-0 feature lines (both applied to Z, per reference)
    run_line(stream, Zin, 32768, lnW[0], lnb[0], lng[0], lnbeta[0],
             HL, MEAN, RSTD, XZ, 512, 0);
    run_line(stream, Zin, 32768, leW[0], leb[0], leg[0], lebeta[0],
             HL, MEAN, RSTD, XZ, 512, 384);

    const float* Xc = Xin;
    const float* Zc = Zin;
    const float* Ac = Ain;
    float* XPbuf[2] = { XP0, XP1 };
    float* ZPbuf[2] = { ZP0, ZP1 };
    float* oS[2]    = { oS0, oS1 };
    const float invScale = 0.070710678118654752f;   // 1/sqrt(200)
    int n = 512;

    for (int i = 0; i < 2; ++i) {
        const int ko   = (i == 0) ? 100 : 50;
        const int rows = 64 * n;

        // Q = X Wq^T + bq ; K = Z Wk^T + bk    (bf16 WMMA, fp32 acc)
        gemm(stream, 0, Xc, Wq[i], Qb, bq[i], 1.f, rows, 200, 200,
             200, 1, 1, 200, 200, 0, 0, 0, 1);
        gemm(stream, 0, Zc, Wk[i], Kb, bk[i], 1.f, rows, 200, 200,
             200, 1, 1, 200, 200, 0, 0, 0, 1);
        // scores = Q K^T / sqrt(f), batched per graph
        gemm(stream, 0, Qb, Kb, SC, nullptr, invScale, n, n, 200,
             200, 1, 1, 200, n, n * 200, n * 200, n * n, 64);
        softmax_mul<<<dim3(rows), 256, 0, stream>>>(SC, Ac, AN, n);
        // X1 = A X
        gemm(stream, 0, AN, Xc, X1b, nullptr, 1.f, n, 200, n,
             n, 1, 200, 1, 200, n * n, n * 200, n * 200, 64);
        // S = X1 w_s  -> straight into d_out
        gemm(stream, 0, X1b, wsP[i], oS[i], nullptr, 1.f, rows, ko, 200,
             200, 1, ko, 1, ko, 0, 0, 0, 1);
        softmax_ent<<<dim3(rows), 128, 0, stream>>>(oS[i], SSM, ENTR, ko);
        // Xp = s^T X ; Zp = s^T Z   (A operand is transposed s: sAm=1,sAk=ko)
        gemm(stream, 0, SSM, Xc, XPbuf[i], nullptr, 1.f, ko, 200, n,
             1, ko, 200, 1, 200, n * ko, n * 200, ko * 200, 64);
        gemm(stream, 0, SSM, Zc, ZPbuf[i], nullptr, 1.f, ko, 200, n,
             1, ko, 200, 1, 200, n * ko, n * 200, ko * 200, 64);
        // As = A s ; Ap = s^T As (= next A; layer 1 writes final_A directly)
        gemm(stream, 0, AN, SSM, ASb, nullptr, 1.f, n, ko, n,
             n, 1, ko, 1, ko, n * n, n * ko, n * ko, 64);
        float* APdst = (i == 0) ? APb : oFA;
        gemm(stream, 0, SSM, ASb, APdst, nullptr, 1.f, ko, ko, n,
             1, ko, ko, 1, ko, n * ko, n * ko, ko * ko, 64);
        // G = s^T s  (for ||s s^T||_F^2 = ||G||_F^2, avoids n x n materialize)
        gemm(stream, 0, SSM, SSM, Gb, nullptr, 1.f, ko, ko, n,
             1, ko, ko, 1, ko, n * ko, n * ko, ko * ko, 64);
        reduce_sq<<<256, 256, 0, stream>>>(AN, 64L * n * n, PAq);
        reduce_sq<<<256, 256, 0, stream>>>(Gb, 64L * ko * ko, PGq);
        finalize_layer<<<1, 256, 0, stream>>>(PAq, PGq, APdst, 64, ko,
            (long)ko * ko, ENTR, (long)rows, (float)(64.0 * n * n), SCAL);

        // GCN over (bz,bz) batch graph: Zc = adj @ Zp_flat, then leaky linear
        gemm(stream, 0, adjp, ZPbuf[i], ZCb, nullptr, 1.f, 64, ko * 200, 64,
             64, 1, ko * 200, 1, ko * 200, 0, 0, 0, 1);
        gemm(stream, 2, ZCb, gcnW, ZPbuf[i], gcnb, 1.f, 64 * ko, 200, 200,
             200, 1, 1, 200, 200, 0, 0, 0, 1);

        // feature lines into padded XZ channels
        run_line(stream, XPbuf[i], 64 * ko, lnW[i + 1], lnb[i + 1],
                 lng[i + 1], lnbeta[i + 1], HL, MEAN, RSTD, XZ, ko, 128 * (i + 1));
        run_line(stream, ZPbuf[i], 64 * ko, leW[i + 1], leb[i + 1],
                 leg[i + 1], lebeta[i + 1], HL, MEAN, RSTD, XZ, ko, 384 + 128 * (i + 1));

        Xc = XPbuf[i]; Zc = ZPbuf[i]; Ac = APdst; n = ko;
    }

    write_scalars<<<1, 32, 0, stream>>>(SCAL, oLink, oEnt);

    // classifier MLP: 768 -> 1024 (relu) -> 128
    gemm(stream, 1, XZ, W1, HC, b1, 1.f, 32768, 1024, 768,
         768, 1, 1, 768, 1024, 0, 0, 0, 1);
    gemm(stream, 0, HC, W2, oY, b2, 1.f, 32768, 128, 1024,
         1024, 1, 1, 1024, 128, 0, 0, 0, 1);
}
